// RingFlashSelfAttention_56495999811611
// MI455X (gfx1250) — matrix operand order repaired and checked by reading, not compiled, not run on hardware
//
#include <hip/hip_runtime.h>

typedef __attribute__((ext_vector_type(16))) _Float16     v16h;
typedef __attribute__((ext_vector_type(8)))  _Float16     v8h;
typedef __attribute__((ext_vector_type(4)))  _Float16     v4h;
typedef __attribute__((ext_vector_type(8)))  float        v8f;
typedef __attribute__((ext_vector_type(4)))  unsigned int v4u;
typedef __attribute__((ext_vector_type(8)))  int          v8i;
typedef __attribute__((ext_vector_type(4)))  int          v4i;

#define SQ 4096
#define NH 16
#define HD 64
#define SCALE 0.125f   // 1/sqrt(64)
#define NEG_BIG (-1.0e30f)

__device__ __forceinline__ v8f wmma_f16(v16h a, v16h b, v8f c) {
  return __builtin_amdgcn_wmma_f32_16x16x32_f16(false, a, false, b, (short)0, c,
                                                false, false);
}

__device__ __forceinline__ v16h cat8(v8h lo, v8h hi) {
  return __builtin_shufflevector(lo, hi, 0, 1, 2, 3, 4, 5, 6, 7,
                                 8, 9, 10, 11, 12, 13, 14, 15);
}

// Flat generic pointer to __shared__: low 32 bits are the LDS byte address
// (ISA 10.2: LDS aperture keeps the LDS offset in addr[31:0]).
__device__ __forceinline__ unsigned int lds_addr_of(const void* p) {
  return (unsigned int)(uintptr_t)p;
}

// ----------------------------------------------------------------------------
// TDM 2D tile load: build Tensor DMA Descriptor per CDNA5 ISA
// §8.3/§8.4 and issue TENSOR_LOAD_TO_LDS. data_size = 2 bytes (f16).
//   tile  : tile_d0 elements per row x tile_d1 rows
//   tensor: tensor_d0 x tensor_d1 (OOB bounds), row stride = stride0 elements
// Tracked by TENSORcnt; EXEC ignored (issue from one wave only).
// This toolchain's builtin takes 6 args (clang-23 form):
//   (uint32x4 g0, int32x8 g1, int32x4 g2, int32x4 g3, int32x8, i32 cpol)
// ----------------------------------------------------------------------------
__device__ __forceinline__ void tdm_load_2d(unsigned int lds_byte_addr,
                                            const void* gaddr,
                                            unsigned int tensor_d0,
                                            unsigned int tensor_d1,
                                            unsigned int tile_d0,
                                            unsigned int tile_d1,
                                            unsigned long long stride0) {
  const unsigned long long ga = (unsigned long long)(uintptr_t)gaddr;
  v4u g0;
  g0[0] = 1u;                                        // count=1 (valid), user mode
  g0[1] = lds_byte_addr;                             // D#.lds_addr
  g0[2] = (unsigned int)ga;                          // global_addr[31:0]
  g0[3] = (unsigned int)((ga >> 32) & 0x01FFFFFFu)   // global_addr[56:32]
          | (2u << 30);                              // type = 2 ("image")
  v8i g1;
  g1[0] = (int)(1u << 16);                           // data_size=1 -> 2 bytes
  g1[1] = (int)((tensor_d0 & 0xFFFFu) << 16);        // tensor_dim0[15:0] @bits63:48
  g1[2] = (int)(((tensor_d0 >> 16) & 0xFFFFu)        // tensor_dim0[31:16]
                | ((tensor_d1 & 0xFFFFu) << 16));    // tensor_dim1[15:0]
  g1[3] = (int)(((tensor_d1 >> 16) & 0xFFFFu)        // tensor_dim1[31:16]
                | ((tile_d0 & 0xFFFFu) << 16));      // tile_dim0
  g1[4] = (int)(tile_d1 & 0xFFFFu);                  // tile_dim1 (tile_dim2 = 0)
  g1[5] = (int)(unsigned int)(stride0 & 0xFFFFFFFFull);        // dim0_stride[31:0]
  g1[6] = (int)(unsigned int)((stride0 >> 32) & 0xFFFFull);    // dim0_stride[47:32]
  g1[7] = 0;                                         // dim1_stride (unused, 2D)
  const v4i z4 = {0, 0, 0, 0};                       // groups 2/3: unused (<=2D)
  const v8i z8 = {0, 0, 0, 0, 0, 0, 0, 0};
  __builtin_amdgcn_tensor_load_to_lds(g0, g1, z4, z4, z8, 0);
}

// ============================================================================
// Pass 1: pack QKV(f32, (s,3,h,d)) -> Q f16 (h,s,d) pre-scaled by 1/sqrt(d),
//         K f16 (h,s,d), V^T f16 (h,d,s) (keys contiguous for B-operands).
// ============================================================================
__global__ __launch_bounds__(256)
void cvt_pack_kernel(const float* __restrict__ qkv,
                     _Float16* __restrict__ qf,
                     _Float16* __restrict__ kf,
                     _Float16* __restrict__ vtf) {
  __shared__ _Float16 vt[64][68];

  const int t  = threadIdx.x;
  const int h  = blockIdx.y;
  const int sb = blockIdx.x;
  const int qd = (t & 15) * 4;

#pragma unroll
  for (int rep = 0; rep < 4; ++rep) {
    const int row = rep * 16 + (t >> 4);
    const size_t s = (size_t)sb * 64 + row;
    const float4 qv = *(const float4*)&qkv[((s * 3 + 0) * NH + h) * HD + qd];
    const float4 kv = *(const float4*)&qkv[((s * 3 + 1) * NH + h) * HD + qd];
    const float4 vv = *(const float4*)&qkv[((s * 3 + 2) * NH + h) * HD + qd];
    v4h qh, kh;
    qh[0] = (_Float16)(qv.x * SCALE); qh[1] = (_Float16)(qv.y * SCALE);
    qh[2] = (_Float16)(qv.z * SCALE); qh[3] = (_Float16)(qv.w * SCALE);
    kh[0] = (_Float16)kv.x; kh[1] = (_Float16)kv.y;
    kh[2] = (_Float16)kv.z; kh[3] = (_Float16)kv.w;
    *(v4h*)&qf[((size_t)h * SQ + s) * HD + qd] = qh;
    *(v4h*)&kf[((size_t)h * SQ + s) * HD + qd] = kh;
    vt[qd + 0][row] = (_Float16)vv.x;
    vt[qd + 1][row] = (_Float16)vv.y;
    vt[qd + 2][row] = (_Float16)vv.z;
    vt[qd + 3][row] = (_Float16)vv.w;
  }
  __syncthreads();
#pragma unroll
  for (int rep = 0; rep < 4; ++rep) {
    const int d  = rep * 16 + (t >> 4);
    const int sq = (t & 15) * 4;
    const v4h o = *(const v4h*)&vt[d][sq];
    *(v4h*)&vtf[((size_t)h * HD + d) * SQ + (size_t)sb * 64 + sq] = o;
  }
}

// ============================================================================
// Pass 2: flash attention. TDM double-buffers the 32-key K/V tiles into LDS
// (one DMA pair per block, issued by wave 0) while all 4 waves run WMMA on the
// previous tile from LDS. Uniform trip count per block -> legal __syncthreads.
// ============================================================================
__global__ __launch_bounds__(128)
void fa_fwd_tdm_kernel(const _Float16* __restrict__ qf,
                       const _Float16* __restrict__ kf,
                       const _Float16* __restrict__ vtf,
                       float* __restrict__ out) {
  __shared__ _Float16 kbuf[2][32][64];   // K tile: key-major (rows s, cols d)
  __shared__ _Float16 vbuf[2][64][32];   // V^T tile: d-major (rows d, cols s)
  __shared__ _Float16 pbuf[4][16][32];   // per-wave P staging

  const int lane  = threadIdx.x & 31;
  const int wave  = threadIdx.x >> 5;
  const int h     = blockIdx.y;
  const int q0blk = blockIdx.x * 64;
  const int q0    = q0blk + wave * 16;

  const int n  = lane & 15;
  const int hi = lane >> 4;

  const _Float16* qh = qf  + (size_t)h * SQ * HD;
  const _Float16* kh = kf  + (size_t)h * SQ * HD;
  const _Float16* vh = vtf + (size_t)h * HD * SQ;

  // Q A-tiles (16x32): lane row = lane%16, K sub-range select by lane-half.
  const int arow = n;
  const int koff = hi * 16;
  const _Float16* qrow = qh + (size_t)(q0 + arow) * HD;
  const v16h aq0 = cat8(*(const v8h*)(qrow + koff),      *(const v8h*)(qrow + 8 + koff));
  const v16h aq1 = cat8(*(const v8h*)(qrow + 32 + koff), *(const v8h*)(qrow + 40 + koff));

  float mrun[8], lrun[8];
#pragma unroll
  for (int r = 0; r < 8; ++r) { mrun[r] = NEG_BIG; lrun[r] = 0.0f; }
  v8f acc[4] = {};

  const int kbmax   = (q0blk + 63) >> 5;   // block-uniform causal bound
  const int mykbmax = (q0 + 15) >> 5;      // this wave's causal bound

  if (wave == 0) {                          // prologue: DMA block 0 into buf 0
    tdm_load_2d(lds_addr_of(&kbuf[0][0][0]), kh, HD, SQ, HD, 32, HD);
    tdm_load_2d(lds_addr_of(&vbuf[0][0][0]), vh, SQ, HD, 32, HD, SQ);
  }

  for (int kb = 0; kb <= kbmax; ++kb) {
    const int nb  = kb * 32;
    const int cur = kb & 1;

    if (wave == 0)
      __builtin_amdgcn_s_wait_tensorcnt(0);   // tile kb landed in LDS
    __syncthreads();                          // publish to all waves; also proves
                                              // buf[cur^1] readers (kb-1) are done
    if (wave == 0 && kb < kbmax) {            // DMA kb+1 while computing kb
      const int nb1 = nb + 32;
      tdm_load_2d(lds_addr_of(&kbuf[cur ^ 1][0][0]), kh + (size_t)nb1 * HD,
                  HD, SQ, HD, 32, HD);
      tdm_load_2d(lds_addr_of(&vbuf[cur ^ 1][0][0]), vh + nb1,
                  SQ, HD, 32, HD, SQ);
    }

    if (kb <= mykbmax) {
      // ================= S = Q * K^T (two 16-key tiles, from LDS) ============
      v8f s[2];
#pragma unroll
      for (int t = 0; t < 2; ++t) {
        const _Float16* krow = &kbuf[cur][t * 16 + n][hi * 16];
        const v16h b0 = *(const v16h*)(krow);        // d = 0..31
        const v16h b1 = *(const v16h*)(krow + 32);   // d = 32..63
        v8f z = {};
        z = wmma_f16(aq0, b0, z);
        z = wmma_f16(aq1, b1, z);
        if (kb < mykbmax) {                  // interior block: no masking needed
#pragma unroll
          for (int r = 0; r < 8; ++r) s[t][r] = z[r];
        } else {                             // diagonal block: causal mask
          const int ki = nb + t * 16 + n;
#pragma unroll
          for (int r = 0; r < 8; ++r)
            s[t][r] = (ki <= q0 + r + 8 * hi) ? z[r] : NEG_BIG;
        }
      }

      // ================= online softmax update ==============================
#pragma unroll
      for (int r = 0; r < 8; ++r) {
        float rm = fmaxf(s[0][r], s[1][r]);
#pragma unroll
        for (int m = 1; m < 16; m <<= 1)     // 16-lane-half butterfly = one row
          rm = fmaxf(rm, __shfl_xor(rm, m, 32));
        const float mnew  = fmaxf(mrun[r], rm);
        const float alpha = __expf(mrun[r] - mnew);
        mrun[r] = mnew;
        const float p0 = __expf(s[0][r] - mnew);
        const float p1 = __expf(s[1][r] - mnew);
        float rs = p0 + p1;
#pragma unroll
        for (int m = 1; m < 16; m <<= 1)
          rs += __shfl_xor(rs, m, 32);
        lrun[r] = lrun[r] * alpha + rs;
#pragma unroll
        for (int db = 0; db < 4; ++db) acc[db][r] *= alpha;
        pbuf[wave][r + 8 * hi][n]      = (_Float16)p0;
        pbuf[wave][r + 8 * hi][n + 16] = (_Float16)p1;
      }

      // wave-local LDS write->read ordering
      asm volatile("s_wait_dscnt 0" ::: "memory");
      const v16h ap = cat8(*(const v8h*)&pbuf[wave][arow][koff],
                           *(const v8h*)&pbuf[wave][arow][8 + koff]);

      // ================= O += P * V (V^T tile from LDS) ======================
#pragma unroll
      for (int db = 0; db < 4; ++db) {
        const int col = db * 16 + n;
        const v16h bv = *(const v16h*)&vbuf[cur][col][hi * 16];
        acc[db] = wmma_f16(ap, bv, acc[db]);
      }
    }
  }

  // ---- epilogue: normalize and store (out is (S, H, D))
#pragma unroll
  for (int r = 0; r < 8; ++r) {
    const float inv = 1.0f / lrun[r];
    const int srec  = q0 + r + 8 * hi;
    float* orow = out + ((size_t)srec * NH + h) * HD;
#pragma unroll
    for (int db = 0; db < 4; ++db)
      orow[db * 16 + n] = acc[db][r] * inv;
  }
}

// ============================================================================
// Fallback (workspace too small): single-pass, converts f32->f16 on the fly.
// ============================================================================
__global__ __launch_bounds__(128)
void fa_fwd_f32_kernel(const float* __restrict__ qkv, float* __restrict__ out) {
  __shared__ _Float16 pbuf[4][16][32];

  const int lane = threadIdx.x & 31;
  const int wave = threadIdx.x >> 5;
  const int h    = blockIdx.y;
  const int q0   = blockIdx.x * 64 + wave * 16;
  const int n  = lane & 15;
  const int hi = lane >> 4;

  const size_t srow = (size_t)3 * NH * HD;
  const float* qbase = qkv + (size_t)h * HD;
  const float* kbase = qkv + (size_t)(NH + h) * HD;
  const float* vbase = qkv + (size_t)(2 * NH + h) * HD;

  const int arow = n;
  const int koff = hi * 16;
  const float* qrow = qbase + (size_t)(q0 + arow) * srow;
  v16h aq0, aq1;
#pragma unroll
  for (int c = 0; c < 2; ++c)
#pragma unroll
    for (int j = 0; j < 8; ++j) {
      aq0[c * 8 + j] = (_Float16)(qrow[c * 8 + koff + j] * SCALE);
      aq1[c * 8 + j] = (_Float16)(qrow[32 + c * 8 + koff + j] * SCALE);
    }

  float mrun[8], lrun[8];
#pragma unroll
  for (int r = 0; r < 8; ++r) { mrun[r] = NEG_BIG; lrun[r] = 0.0f; }
  v8f acc[4] = {};

  const int kbmax = (q0 + 15) >> 5;
  for (int kb = 0; kb <= kbmax; ++kb) {
    const int nb = kb * 32;
    v8f s[2];
#pragma unroll
    for (int t = 0; t < 2; ++t) {
      const int n0t = nb + t * 16;
      const float* krow = kbase + (size_t)(n0t + n) * srow;
      v16h b0, b1;
#pragma unroll
      for (int e = 0; e < 16; ++e) {
        b0[e] = (_Float16)krow[hi * 16 + e];
        b1[e] = (_Float16)krow[32 + hi * 16 + e];
      }
      v8f z = {};
      z = wmma_f16(aq0, b0, z);
      z = wmma_f16(aq1, b1, z);
#pragma unroll
      for (int r = 0; r < 8; ++r)
        s[t][r] = ((n0t + n) <= q0 + r + 8 * hi) ? z[r] : NEG_BIG;
    }
#pragma unroll
    for (int r = 0; r < 8; ++r) {
      float rm = fmaxf(s[0][r], s[1][r]);
#pragma unroll
      for (int m = 1; m < 16; m <<= 1) rm = fmaxf(rm, __shfl_xor(rm, m, 32));
      const float mnew  = fmaxf(mrun[r], rm);
      const float alpha = __expf(mrun[r] - mnew);
      mrun[r] = mnew;
      const float p0 = __expf(s[0][r] - mnew);
      const float p1 = __expf(s[1][r] - mnew);
      float rs = p0 + p1;
#pragma unroll
      for (int m = 1; m < 16; m <<= 1) rs += __shfl_xor(rs, m, 32);
      lrun[r] = lrun[r] * alpha + rs;
#pragma unroll
      for (int db = 0; db < 4; ++db) acc[db][r] *= alpha;
      pbuf[wave][r + 8 * hi][n]      = (_Float16)p0;
      pbuf[wave][r + 8 * hi][n + 16] = (_Float16)p1;
    }
    asm volatile("s_wait_dscnt 0" ::: "memory");
    const v16h ap = cat8(*(const v8h*)&pbuf[wave][arow][koff],
                         *(const v8h*)&pbuf[wave][arow][8 + koff]);
#pragma unroll
    for (int db = 0; db < 4; ++db) {
      const int col = db * 16 + n;
      const float* vcol = vbase + col;
      v16h bv;
#pragma unroll
      for (int e = 0; e < 16; ++e)
        bv[e] = (_Float16)vcol[(size_t)(nb + hi * 16 + e) * srow];
      acc[db] = wmma_f16(ap, bv, acc[db]);
    }
  }
#pragma unroll
  for (int r = 0; r < 8; ++r) {
    const float inv = 1.0f / lrun[r];
    const int srec  = q0 + r + 8 * hi;
    float* orow = out + ((size_t)srec * NH + h) * HD;
#pragma unroll
    for (int db = 0; db < 4; ++db)
      orow[db * 16 + n] = acc[db][r] * inv;
  }
}

extern "C" void kernel_launch(void* const* d_in, const int* in_sizes, int n_in,
                              void* d_out, int out_size, void* d_ws, size_t ws_size,
                              hipStream_t stream) {
  (void)in_sizes; (void)n_in; (void)out_size;
  const float* qkv = (const float*)d_in[0];
  float* out = (float*)d_out;

  const size_t elems = (size_t)SQ * NH * HD;
  const size_t need  = 3 * elems * sizeof(_Float16);   // 24 MiB

  if (ws_size >= need) {
    _Float16* qf  = (_Float16*)d_ws;
    _Float16* kf  = qf + elems;
    _Float16* vtf = kf + elems;
    cvt_pack_kernel<<<dim3(SQ / 64, NH), 256, 0, stream>>>(qkv, qf, kf, vtf);
    fa_fwd_tdm_kernel<<<dim3(SQ / 64, NH), 128, 0, stream>>>(qf, kf, vtf, out);
  } else {
    fa_fwd_f32_kernel<<<dim3(SQ / 64, NH), 128, 0, stream>>>(qkv, out);
  }
}
